// CELPNet_48765058679236
// MI455X (gfx1250) — matrix operand order repaired and checked by reading, not compiled, not run on hardware
//
#include <hip/hip_runtime.h>
#include <hip/hip_bf16.h>
#include <math.h>

// ---------------------------------------------------------------------------
// CELPNet vocoder for MI455X (gfx1250, wave32, WMMA bf16 16x16x32).
//  - conditioning net:  generic WMMA GEMM kernel (conv1d via contiguous im2col)
//  - sequential scan:   persistent kernel, 8 blocks x 256 thr (64 waves),
//                       grid barrier between dependent stages, double-buffered
//                       GRU hidden state, gate math fused on WMMA accumulators.
// Weights are pre-swizzled to the CDNA5 B-operand lane layout so each lane
// loads its 16 K-elements with one 32B load; activations stay bf16 row-major
// so the A-operand is two 16B loads per lane (ISA 16-bit A 16x32 layout).
// ---------------------------------------------------------------------------

typedef __attribute__((ext_vector_type(16))) __bf16 v16bf;
typedef __attribute__((ext_vector_type(8)))  __bf16 v8bf;
typedef __attribute__((ext_vector_type(8)))  float  v8f;

#define NBLK 8          // blocks in persistent scan kernel (must be co-resident)
#define T_   400        // nb_frames * NSUB
#define SUB_ 40

// ---------------- WMMA helpers --------------------------------------------

__device__ __forceinline__ v16bf load_a32(const __bf16* __restrict__ row, int kt, int koff) {
  // 16-bit A 16x32 layout: lane<16 -> K {kb..kb+7, kb+16..kb+23},
  //                        lane>=16 -> K {kb+8..kb+15, kb+24..kb+31}; koff=(lane>>4)*8
  const __bf16* p = row + kt * 32 + koff;
  v8bf lo = *(const v8bf*)(p);
  v8bf hi = *(const v8bf*)(p + 16);
  v16bf a;
#pragma unroll
  for (int i = 0; i < 8; ++i) { a[i] = lo[i]; a[i + 8] = hi[i]; }
  return a;
}

__device__ __forceinline__ v16bf load_b32(const __bf16* __restrict__ Wt, int nt, int kt,
                                          int KT, int lane) {
  // pre-swizzled: tile (nt,kt), per-lane 16 contiguous bf16 (one 32B load)
  return *(const v16bf*)(Wt + (((size_t)nt * KT + kt) * 32 + lane) * 16);
}

__device__ __forceinline__ v8f wmma_bf16(v16bf a, v16bf b, v8f c) {
  return __builtin_amdgcn_wmma_f32_16x16x32_bf16(false, a, false, b, (short)0, c,
                                                 false, false);
}

__device__ __forceinline__ v8f splat8(float x) {
  v8f r;
#pragma unroll
  for (int i = 0; i < 8; ++i) r[i] = x;
  return r;
}

__device__ __forceinline__ float sigf(float x) { return 1.0f / (1.0f + __expf(-x)); }

// ---------------- grid barrier (sense via generation counter) -------------

__device__ __forceinline__ void grid_sync(int* cnt, int* gen) {
  __syncthreads();
  if (threadIdx.x == 0) {
    __threadfence();
    int g = __atomic_load_n(gen, __ATOMIC_RELAXED);
    if (__atomic_fetch_add(cnt, 1, __ATOMIC_ACQ_REL) == NBLK - 1) {
      __atomic_store_n(cnt, 0, __ATOMIC_RELAXED);
      __atomic_store_n(gen, g + 1, __ATOMIC_RELEASE);
    } else {
      while (__atomic_load_n(gen, __ATOMIC_ACQUIRE) == g)
        __builtin_amdgcn_s_sleep(2);
    }
    __threadfence();
  }
  __syncthreads();
}

// ---------------- weight swizzle: f32 [N,K] -> tiled bf16 -----------------
// out[(((nt*KT+kt)*32+lane)*16)+e] = W[nt*16+lane%16, kt*32+(e/8)*16+(lane/16)*8+e%8]
// conv==1: W is torch Conv1d [O,I,3]; flat K index kk maps to (i=kk%256, k3=kk/256)
// so that im2col rows are contiguous in a [B,L,256] activation buffer.

__global__ void k_prep_w(const float* __restrict__ W, __bf16* __restrict__ Out,
                         int N, int K, int Kp, int conv, int total) {
  const int KT = Kp >> 5;
  for (int i = blockIdx.x * blockDim.x + threadIdx.x; i < total;
       i += gridDim.x * blockDim.x) {
    int e = i & 15;
    int lane = (i >> 4) & 31;
    int tileIdx = i >> 9;
    int kt = tileIdx % KT;
    int nt = tileIdx / KT;
    int n = nt * 16 + (lane & 15);
    int k = kt * 32 + ((e >> 3) << 4) + ((lane >> 4) << 3) + (e & 7);
    float v = 0.f;
    if (n < N && k < K)
      v = conv ? W[((size_t)n * 256 + (k & 255)) * 3 + (k >> 8)]
               : W[(size_t)n * K + k];
    Out[i] = (__bf16)v;
  }
}

// ---------------- build fd1 input: [64*104, 96] bf16 ----------------------

__global__ void k_prep_f(const float* __restrict__ features, const int* __restrict__ period,
                         const float* __restrict__ pembed, __bf16* __restrict__ fb,
                         int total) {
  for (int i = blockIdx.x * blockDim.x + threadIdx.x; i < total;
       i += gridDim.x * blockDim.x) {
    int r = i / 96, c = i - r * 96;
    float v = 0.f;
    if (c < 20)       v = features[(size_t)r * 20 + c];
    else if (c < 84)  v = pembed[(size_t)period[r] * 64 + (c - 20)];
    fb[i] = (__bf16)v;
  }
}

// ---------------- phase embedding -> pr/pi bf16 [400,64,40] ---------------

__global__ void k_phase(const int* __restrict__ period, __bf16* __restrict__ pr,
                        __bf16* __restrict__ pi) {
  __shared__ float cum[100], w0s[100];
  const int b = blockIdx.x;
  if (threadIdx.x == 0) {
    float s = 0.f, wprev = 0.f;
    for (int j = 0; j < 100; ++j) {
      float w0 = 6.283185307179586f / (float)period[b * 104 + 3 + j];
      float ws = (j == 0) ? 0.f : wprev;   // random shift approximated as 0
      s += ws;
      cum[j] = 160.f * s;
      w0s[j] = w0;
      wprev = w0;
    }
  }
  __syncthreads();
  for (int g = threadIdx.x; g < 16000; g += blockDim.x) {
    int j = g / 160, ii = g - j * 160;
    float e = cum[j] + w0s[j] * (float)ii;
    int tt = g / 40, s2 = g - tt * 40;
    size_t idx = ((size_t)tt * 64 + b) * 40 + s2;
    pr[idx] = (__bf16)cosf(e);
    pi[idx] = (__bf16)sinf(e);
  }
}

// ---------------- generic GEMM + tanh (conditioning net) ------------------
// out[M,256] = tanh(A @ W^T + bias); A row r maps to base (r/Lout*Lin + r%Lout)
// * rowStride (conv1d im2col: K=768 spans 3 contiguous rows).

__global__ void __launch_bounds__(256) k_gemm_tanh(
    const __bf16* __restrict__ A, const __bf16* __restrict__ Wt,
    const float* __restrict__ bias, __bf16* __restrict__ Out,
    int M, int K, int Lin, int Lout, int rowStride) {
  const int lane = threadIdx.x & 31;
  const int gw = blockIdx.x * 8 + (threadIdx.x >> 5);
  const int nwaves = gridDim.x * 8;
  const int KT = K >> 5;
  const int ntiles = (M >> 4) << 4;   // (M/16) * (256/16)
  const int koff = (lane >> 4) << 3;
  for (int tile = gw; tile < ntiles; tile += nwaves) {
    const int rt = tile >> 4, ct = tile & 15;
    const int r = rt * 16 + (lane & 15);
    const int b = r / Lout, l = r - b * Lout;
    const __bf16* arow = A + (size_t)(b * Lin + l) * rowStride;
    v8f acc = splat8(bias[ct * 16 + (lane & 15)]);
    for (int kt = 0; kt < KT; ++kt)
      acc = wmma_bf16(load_a32(arow, kt, koff), load_b32(Wt, ct, kt, KT, lane), acc);
    const int n = ct * 16 + (lane & 15);
    const int mb = rt * 16 + koff;      // D layout: M = v + 8*(lane>=16)
#pragma unroll
    for (int v = 0; v < 8; ++v)
      Out[(size_t)(mb + v) * 256 + n] = (__bf16)tanhf(acc[v]);
  }
}

// ---------------- persistent sequential scan ------------------------------

__global__ void __launch_bounds__(256) k_scan(
    const __bf16* __restrict__ Wsd1, const __bf16* __restrict__ Wsd2,
    const __bf16* __restrict__ Wg1i, const __bf16* __restrict__ Wg1h,
    const __bf16* __restrict__ Wg2i, const __bf16* __restrict__ Wg2h,
    const __bf16* __restrict__ Wg3i, const __bf16* __restrict__ Wg3h,
    const __bf16* __restrict__ Wsdo,
    const float* __restrict__ sd1_b, const float* __restrict__ sd2_b,
    const float* __restrict__ b1i, const float* __restrict__ b1h,
    const float* __restrict__ b2i, const float* __restrict__ b2h,
    const float* __restrict__ b3i, const float* __restrict__ b3h,
    const float* __restrict__ sdo_b,
    const __bf16* __restrict__ cond, const __bf16* __restrict__ pr,
    const __bf16* __restrict__ pi,
    __bf16* __restrict__ xbuf, __bf16* __restrict__ a1, __bf16* __restrict__ a2,
    __bf16* __restrict__ hb, float* __restrict__ hf,
    float* __restrict__ dout, int* __restrict__ bar) {
  const int tid = threadIdx.x;
  const int lane = tid & 31;
  const int gw = blockIdx.x * 8 + (tid >> 5);      // 0..63
  const int koff = (lane >> 4) << 3;
  int* cnt = bar;
  int* gen = bar + 1;

  const __bf16* Wgi[3] = {Wg1i, Wg2i, Wg3i};
  const __bf16* Wgh[3] = {Wg1h, Wg2h, Wg3h};
  const float*  bgi[3] = {b1i, b2i, b3i};
  const float*  bgh[3] = {b1h, b2h, b3h};

  // ---- init: zero double-buffered h, build x(t=0) ----
  for (int i = blockIdx.x * 256 + tid; i < 6 * 16384; i += NBLK * 256) {
    hf[i] = 0.f;
    hb[i] = (__bf16)0.f;
  }
  for (int i = blockIdx.x * 256 + tid; i < 64 * 384; i += NBLK * 256) {
    int b = i / 384, c = i - b * 384;
    float v = 0.f;
    if (c < 256)      v = (float)cond[(size_t)(b * 100) * 256 + c];
    else if (c < 296) v = 0.f;                                   // prev output
    else if (c < 336) v = (float)pr[(size_t)b * 40 + (c - 296)];
    else if (c < 376) v = (float)pi[(size_t)b * 40 + (c - 336)];
    xbuf[i] = (__bf16)v;                                         // cols 376..383 = 0 pad
  }
  grid_sync(cnt, gen);

  for (int t = 0; t < T_; ++t) {
    const int cur = t & 1, nxt = cur ^ 1;

    // ---- stage 1: a1 = tanh(x @ sd1_w^T + b), K=384 (padded 376) ----
    {
      const int rt = gw >> 4, ct = gw & 15;
      const __bf16* arow = xbuf + (size_t)(rt * 16 + (lane & 15)) * 384;
      v8f acc = splat8(sd1_b[ct * 16 + (lane & 15)]);
#pragma unroll
      for (int kt = 0; kt < 12; ++kt)
        acc = wmma_bf16(load_a32(arow, kt, koff), load_b32(Wsd1, ct, kt, 12, lane), acc);
      const int n = ct * 16 + (lane & 15);
      const int mb = rt * 16 + koff;
#pragma unroll
      for (int v = 0; v < 8; ++v)
        a1[(size_t)(mb + v) * 256 + n] = (__bf16)tanhf(acc[v]);
    }
    grid_sync(cnt, gen);

    // ---- stage 2: a2 = tanh(a1 @ sd2_w^T + b), K=256 ----
    {
      const int rt = gw >> 4, ct = gw & 15;
      const __bf16* arow = a1 + (size_t)(rt * 16 + (lane & 15)) * 256;
      v8f acc = splat8(sd2_b[ct * 16 + (lane & 15)]);
#pragma unroll
      for (int kt = 0; kt < 8; ++kt)
        acc = wmma_bf16(load_a32(arow, kt, koff), load_b32(Wsd2, ct, kt, 8, lane), acc);
      const int n = ct * 16 + (lane & 15);
      const int mb = rt * 16 + koff;
#pragma unroll
      for (int v = 0; v < 8; ++v)
        a2[(size_t)(mb + v) * 256 + n] = (__bf16)tanhf(acc[v]);
    }
    grid_sync(cnt, gen);

    // ---- stages 3..5: GRU chain, gates fused on accumulators ----
#pragma unroll
    for (int g = 0; g < 3; ++g) {
      const __bf16* in  = (g == 0) ? a2 : (hb + (size_t)((g - 1) * 2 + nxt) * 16384);
      const __bf16* hbc = hb + (size_t)(g * 2 + cur) * 16384;
      const float*  hfc = hf + (size_t)(g * 2 + cur) * 16384;
      float*  hfn = hf + (size_t)(g * 2 + nxt) * 16384;
      __bf16* hbn = hb + (size_t)(g * 2 + nxt) * 16384;
      const __bf16* Wi = Wgi[g];
      const __bf16* Wh = Wgh[g];
      const float* bi = bgi[g];
      const float* bh = bgh[g];
      const int rt = gw >> 4, nt = gw & 15;
      const int nn = nt * 16 + (lane & 15);
      v8f gi0 = splat8(bi[nn]),       gh0 = splat8(bh[nn]);
      v8f gi1 = splat8(bi[256 + nn]), gh1 = splat8(bh[256 + nn]);
      v8f gi2 = splat8(bi[512 + nn]), gh2 = splat8(bh[512 + nn]);
      const __bf16* arow_i = in  + (size_t)(rt * 16 + (lane & 15)) * 256;
      const __bf16* arow_h = hbc + (size_t)(rt * 16 + (lane & 15)) * 256;
#pragma unroll
      for (int kt = 0; kt < 8; ++kt) {
        if (kt + 1 < 8)  // stream next weight tiles through L2
          __builtin_prefetch((const void*)(Wi + (((size_t)nt * 8 + kt + 1) * 32 + lane) * 16), 0, 1);
        v16bf ai = load_a32(arow_i, kt, koff);
        v16bf ah = load_a32(arow_h, kt, koff);
        gi0 = wmma_bf16(ai, load_b32(Wi, nt,      kt, 8, lane), gi0);
        gi1 = wmma_bf16(ai, load_b32(Wi, nt + 16, kt, 8, lane), gi1);
        gi2 = wmma_bf16(ai, load_b32(Wi, nt + 32, kt, 8, lane), gi2);
        gh0 = wmma_bf16(ah, load_b32(Wh, nt,      kt, 8, lane), gh0);
        gh1 = wmma_bf16(ah, load_b32(Wh, nt + 16, kt, 8, lane), gh1);
        gh2 = wmma_bf16(ah, load_b32(Wh, nt + 32, kt, 8, lane), gh2);
      }
      const int mb = rt * 16 + koff;
#pragma unroll
      for (int v = 0; v < 8; ++v) {
        const size_t idx = (size_t)(mb + v) * 256 + nn;
        float r  = sigf(gi0[v] + gh0[v]);
        float z  = sigf(gi1[v] + gh1[v]);
        float gg = tanhf(gi2[v] + r * gh2[v]);
        float hn = (1.f - z) * gg + z * hfc[idx];
        hfn[idx] = hn;
        hbn[idx] = (__bf16)hn;
      }
      grid_sync(cnt, gen);
    }

    // ---- stage 6: sdo output (waves 0..11) + x(t+1) prep (waves 12..63) ----
    if (gw < 12) {
      const int rt = gw / 3, ct = gw - rt * 3;          // N padded 40 -> 48
      const __bf16* h3 = hb + (size_t)(4 + nxt) * 16384;
      const __bf16* arow = h3 + (size_t)(rt * 16 + (lane & 15)) * 256;
      const int s0 = ct * 16 + (lane & 15);
      v8f acc = splat8(s0 < SUB_ ? sdo_b[s0] : 0.f);
#pragma unroll
      for (int kt = 0; kt < 8; ++kt)
        acc = wmma_bf16(load_a32(arow, kt, koff), load_b32(Wsdo, ct, kt, 8, lane), acc);
      if (s0 < SUB_) {
        const int mb = rt * 16 + koff;
#pragma unroll
        for (int v = 0; v < 8; ++v) {
          float o = tanhf(acc[v]);
          dout[(size_t)(mb + v) * 16000 + t * 40 + s0] = o;       // sig[b, t*40+s]
          xbuf[(size_t)(mb + v) * 384 + 256 + s0] = (__bf16)o;    // prev for t+1
        }
      }
    } else if (t + 1 < T_) {
      const int hid = (gw - 12) * 32 + lane;
      const int frame = (t + 1) >> 2;
      for (int i = hid; i < 64 * 336; i += 52 * 32) {
        int b = i / 336, c = i - b * 336;
        float v;
        int col;
        if (c < 256)      { v = (float)cond[((size_t)b * 100 + frame) * 256 + c]; col = c; }
        else if (c < 296) { v = (float)pr[((size_t)(t + 1) * 64 + b) * 40 + (c - 256)]; col = c + 40; }
        else              { v = (float)pi[((size_t)(t + 1) * 64 + b) * 40 + (c - 296)]; col = c + 40; }
        xbuf[(size_t)b * 384 + col] = (__bf16)v;
      }
    }
    grid_sync(cnt, gen);
  }
}

// ---------------------------------------------------------------------------

extern "C" void kernel_launch(void* const* d_in, const int* in_sizes, int n_in,
                              void* d_out, int out_size, void* d_ws, size_t ws_size,
                              hipStream_t stream) {
  const float* features = (const float*)d_in[0];
  const int*   period   = (const int*)d_in[1];
  const float* pembed   = (const float*)d_in[3];
  const float* fd1_w = (const float*)d_in[4];  const float* fd1_b = (const float*)d_in[5];
  const float* fc1_w = (const float*)d_in[6];  const float* fc1_b = (const float*)d_in[7];
  const float* fc2_w = (const float*)d_in[8];  const float* fc2_b = (const float*)d_in[9];
  const float* fd2_w = (const float*)d_in[10]; const float* fd2_b = (const float*)d_in[11];
  const float* sd1_w = (const float*)d_in[12]; const float* sd1_b = (const float*)d_in[13];
  const float* sd2_w = (const float*)d_in[14]; const float* sd2_b = (const float*)d_in[15];
  const float* g_wih[3] = {(const float*)d_in[16], (const float*)d_in[20], (const float*)d_in[24]};
  const float* g_whh[3] = {(const float*)d_in[17], (const float*)d_in[21], (const float*)d_in[25]};
  const float* g_bih[3] = {(const float*)d_in[18], (const float*)d_in[22], (const float*)d_in[26]};
  const float* g_bhh[3] = {(const float*)d_in[19], (const float*)d_in[23], (const float*)d_in[27]};
  const float* sdo_w = (const float*)d_in[28]; const float* sdo_b = (const float*)d_in[29];
  float* out = (float*)d_out;

  char* ws = (char*)d_ws;
  size_t cur = 0;
  auto alloc = [&](size_t bytes) -> void* {
    void* p = ws + cur;
    cur = (cur + bytes + 255) & ~(size_t)255;
    return p;
  };

  int* bar = (int*)alloc(256);
  __bf16* Wfd1 = (__bf16*)alloc((size_t)24576  * 2);
  __bf16* Wfc1 = (__bf16*)alloc((size_t)196608 * 2);
  __bf16* Wfc2 = (__bf16*)alloc((size_t)196608 * 2);
  __bf16* Wfd2 = (__bf16*)alloc((size_t)65536  * 2);
  __bf16* Wsd1 = (__bf16*)alloc((size_t)98304  * 2);
  __bf16* Wsd2 = (__bf16*)alloc((size_t)65536  * 2);
  __bf16* Wgi[3]; __bf16* Wgh[3];
  for (int g = 0; g < 3; ++g) {
    Wgi[g] = (__bf16*)alloc((size_t)196608 * 2);
    Wgh[g] = (__bf16*)alloc((size_t)196608 * 2);
  }
  __bf16* Wsdo = (__bf16*)alloc((size_t)12288 * 2);
  __bf16* f_bf = (__bf16*)alloc((size_t)6656 * 96 * 2);
  __bf16* t1   = (__bf16*)alloc((size_t)6656 * 256 * 2);
  __bf16* t2   = (__bf16*)alloc((size_t)6528 * 256 * 2);
  __bf16* t3   = (__bf16*)alloc((size_t)6400 * 256 * 2);
  __bf16* cond = (__bf16*)alloc((size_t)6400 * 256 * 2);
  __bf16* prB  = (__bf16*)alloc((size_t)T_ * 64 * 40 * 2);
  __bf16* piB  = (__bf16*)alloc((size_t)T_ * 64 * 40 * 2);
  __bf16* xbuf = (__bf16*)alloc((size_t)64 * 384 * 2);
  __bf16* a1   = (__bf16*)alloc((size_t)64 * 256 * 2);
  __bf16* a2   = (__bf16*)alloc((size_t)64 * 256 * 2);
  __bf16* hb   = (__bf16*)alloc((size_t)6 * 16384 * 2);   // [gru][buf][64x256]
  float*  hf   = (float*)alloc((size_t)6 * 16384 * 4);

  hipMemsetAsync(bar, 0, 8, stream);

  auto prep_w = [&](const float* W, __bf16* Out, int N, int K, int Kp, int conv) {
    int Np = (N + 15) & ~15;
    int total = (Np / 16) * (Kp / 32) * 512;
    k_prep_w<<<(total + 255) / 256, 256, 0, stream>>>(W, Out, N, K, Kp, conv, total);
  };
  prep_w(fd1_w, Wfd1, 256, 84, 96, 0);
  prep_w(fc1_w, Wfc1, 256, 768, 768, 1);
  prep_w(fc2_w, Wfc2, 256, 768, 768, 1);
  prep_w(fd2_w, Wfd2, 256, 256, 256, 0);
  prep_w(sd1_w, Wsd1, 256, 376, 384, 0);
  prep_w(sd2_w, Wsd2, 256, 256, 256, 0);
  for (int g = 0; g < 3; ++g) {
    prep_w(g_wih[g], Wgi[g], 768, 256, 256, 0);
    prep_w(g_whh[g], Wgh[g], 768, 256, 256, 0);
  }
  prep_w(sdo_w, Wsdo, 40, 256, 256, 0);

  {
    int total = 6656 * 96;
    k_prep_f<<<(total + 255) / 256, 256, 0, stream>>>(features, period, pembed, f_bf, total);
  }
  k_phase<<<64, 256, 0, stream>>>(period, prB, piB);

  auto gemm = [&](const __bf16* A, const __bf16* Wt, const float* bias, __bf16* Out,
                  int M, int K, int Lin, int Lout, int rowStride) {
    int tiles = (M >> 4) * 16;
    int blocks = (tiles + 7) / 8;
    k_gemm_tanh<<<blocks, 256, 0, stream>>>(A, Wt, bias, Out, M, K, Lin, Lout, rowStride);
  };
  gemm(f_bf, Wfd1, fd1_b, t1, 6656, 96,  104, 104, 96);   // fd1
  gemm(t1,   Wfc1, fc1_b, t2, 6528, 768, 104, 102, 256);  // conv1 (im2col rows contiguous)
  gemm(t2,   Wfc2, fc2_b, t3, 6400, 768, 102, 100, 256);  // conv2
  gemm(t3,   Wfd2, fd2_b, cond, 6400, 256, 100, 100, 256);// fd2 -> cond [B,100,256]

  k_scan<<<NBLK, 256, 0, stream>>>(
      Wsd1, Wsd2, Wgi[0], Wgh[0], Wgi[1], Wgh[1], Wgi[2], Wgh[2], Wsdo,
      sd1_b, sd2_b, g_bih[0], g_bhh[0], g_bih[1], g_bhh[1], g_bih[2], g_bhh[2], sdo_b,
      cond, prB, piB, xbuf, a1, a2, hb, hf, out, bar);

  (void)in_sizes; (void)n_in; (void)out_size; (void)ws_size;
}
